// MotionNetwork_44487271252337
// MI455X (gfx1250) — matrix operand order repaired and checked by reading, compiled, not run on hardware
//
#include <hip/hip_runtime.h>
#include <hip/hip_fp16.h>

// ---------------------------------------------------------------------------
// MotionNetwork for MI455X (gfx1250, wave32, WMMA).
//   knn -> gather/pack -> 3x [WMMA GEMM + bias + ReLU] x 2 branches
//       -> per-object final layer (VALU, reads obj on device)
// All activations are stored (N, K) row-major in f16 so that:
//   * WMMA B-tiles stage with ds_store_b128 (no transpose),
//   * GEMM epilogues write contiguous 16B v8h stores,
//   * the final layer reads each point's 128 features contiguously.
// ---------------------------------------------------------------------------

typedef __attribute__((ext_vector_type(16))) _Float16 v16h;
typedef __attribute__((ext_vector_type(8)))  _Float16 v8h;
typedef __attribute__((ext_vector_type(8)))  float    v8f;

constexpr int N_PTS = 8192;
constexpr int F_IN  = 512;
constexpr int C_IN  = 2 * F_IN;   // 1024
constexpr int L1D = 640, L2D = 256, L3D = 128;

// GEMM tiling: block tile 128(M) x 128(N), K stepped by 32 (WMMA K).
constexpr int BM = 128;
constexpr int BN = 128;
constexpr int BK = 32;
constexpr int TSTR = 40;  // padded LDS row stride (halves) for [row][k] tiles

// ---------------------------------------------------------------------------
// 1) Brute-force 1-NN: for each t2 point, argmin distance over all t1 points.
// ---------------------------------------------------------------------------
__global__ __launch_bounds__(256)
void knn_kernel(const float* __restrict__ t1, const float* __restrict__ t2,
                int* __restrict__ idx_out) {
  __shared__ float sx[1024], sy[1024], sz[1024];
  const int q = blockIdx.x * blockDim.x + threadIdx.x;
  const float qx = t2[q];
  const float qy = t2[N_PTS + q];
  const float qz = t2[2 * N_PTS + q];
  float best = 3.402823466e38f;
  int bestI = 0;
  for (int base = 0; base < N_PTS; base += 1024) {
    for (int i = threadIdx.x; i < 1024; i += blockDim.x) {
      sx[i] = t1[base + i];
      sy[i] = t1[N_PTS + base + i];
      sz[i] = t1[2 * N_PTS + base + i];
    }
    __syncthreads();
#pragma unroll 8
    for (int i = 0; i < 1024; ++i) {
      const float dx = sx[i] - qx, dy = sy[i] - qy, dz = sz[i] - qz;
      const float d2 = fmaf(dx, dx, fmaf(dy, dy, dz * dz));
      if (d2 < best) { best = d2; bestI = base + i; }  // first-min like argmin
    }
    __syncthreads();
  }
  idx_out[q] = bestI;
}

// ---------------------------------------------------------------------------
// 2) fp32 -> f16 weight conversion (weights are tiny; done per launch).
// ---------------------------------------------------------------------------
__global__ void cvt_f32_f16_kernel(const float* __restrict__ src,
                                   _Float16* __restrict__ dst, int n) {
  const int i = blockIdx.x * blockDim.x + threadIdx.x;
  if (i < n) dst[i] = (_Float16)src[i];
}

// ---------------------------------------------------------------------------
// 3) Build X (N_PTS x C_IN) f16:  X[n][0..512) = emb1[:, idx[n]],
//    X[n][512..1024) = emb2[:, n].  Tiled 64n x 64f transpose through LDS so
//    emb2 reads and all X writes are coalesced (emb1 reads are a gather and
//    scattered regardless; emb1/emb2 are L2-resident at 33.5 MB total).
// ---------------------------------------------------------------------------
__global__ __launch_bounds__(256)
void pack_x_kernel(const float* __restrict__ emb1, const float* __restrict__ emb2,
                   const int* __restrict__ idx, _Float16* __restrict__ X) {
  __shared__ _Float16 tile[64][65];
  __shared__ int jidx[64];
  const int n0 = blockIdx.x * 64;
  const int f0 = blockIdx.y * 64;
  const int tid = threadIdx.x;
  if (tid < 64) jidx[tid] = idx[n0 + tid];
  __syncthreads();

  {  // read phase: lanes sweep n (coalesced for emb2); 16 f-rows per thread
    const int nl = tid & 63;
    const int fl0 = tid >> 6;  // 0..3
#pragma unroll
    for (int i = 0; i < 16; ++i) {
      const int fl = fl0 + i * 4;
      const int f = f0 + fl;
      float v;
      if (f < F_IN) v = emb1[(size_t)f * N_PTS + jidx[nl]];
      else          v = emb2[(size_t)(f - F_IN) * N_PTS + n0 + nl];
      tile[nl][fl] = (_Float16)v;
    }
  }
  __syncthreads();

  {  // write phase: 8 contiguous halves per thread along f, coalesced
    const int fl0 = (tid & 7) * 8;
    const int nl0 = tid >> 3;  // 0..31
#pragma unroll
    for (int p = 0; p < 2; ++p) {
      const int nl = nl0 + 32 * p;
      v8h o;
#pragma unroll
      for (int i = 0; i < 8; ++i) o[i] = tile[nl][fl0 + i];
      *(v8h*)&X[(size_t)(n0 + nl) * C_IN + f0 + fl0] = o;
    }
  }
}

// ---------------------------------------------------------------------------
// WMMA fragment loads (layouts per CDNA5 ISA 7.12.2, wave32).
// Both tiles live in LDS as [row][k], row stride TSTR halves.
// A (16x32 f16): lane<16: row=lane,    K={0..7}u{16..23}
//                lane>=16: row=lane-16, K={8..15}u{24..31}
// ---------------------------------------------------------------------------
__device__ inline v16h load_frag_a(const _Float16* aTile, int mBase, int lane) {
  const _Float16* p = aTile + (size_t)(mBase + (lane & 15)) * TSTR + ((lane >> 4) << 3);
  union { v16h v; v8h h[2]; } u;
  u.h[0] = *(const v8h*)p;         // K khalf .. khalf+7
  u.h[1] = *(const v8h*)(p + 16);  // K khalf+16 .. khalf+23
  return u.v;
}

// B (32x16 f16): lane<16: col N=nBase+lane,    K={0..7}u{16..23}
//                lane>=16: col N=nBase+lane-16, K={8..15}u{24..31}
__device__ inline v16h load_frag_b(const _Float16* bTile, int nBase, int lane) {
  const _Float16* p = bTile + (size_t)(nBase + (lane & 15)) * TSTR + ((lane >> 4) << 3);
  union { v16h v; v8h h[2]; } u;
  u.h[0] = *(const v8h*)p;
  u.h[1] = *(const v8h*)(p + 16);
  return u.v;
}

// ---------------------------------------------------------------------------
// 4) Y = relu(W @ X^T + bias)^T stored (N, M):
//    W:(M,K) f16 row-major, X:(N_PTS,K) f16 row-major, Y:(N_PTS,M) f16.
//    8 waves (4Mx2N), each wave: 2 M-tiles x 4 N-tiles = 8 WMMA / K-step.
//    Register-staged + double-buffered LDS: one barrier per K-step; the
//    next slice's global loads fly under the matrix pipe. K loop unrolled
//    by 2 so buffer selection is compile-time (nk is always even).
// ---------------------------------------------------------------------------
__global__ __launch_bounds__(256)
void gemm_bias_relu_wmma(const _Float16* __restrict__ W,
                         const _Float16* __restrict__ X,
                         const float* __restrict__ bias,
                         _Float16* __restrict__ Y,
                         int M, int K) {
  __shared__ _Float16 aT[2][BM * TSTR];  // [m][k]
  __shared__ _Float16 bT[2][BN * TSTR];  // [n][k]

  const int tid    = threadIdx.x;
  const int lane   = tid & 31;
  const int wave   = tid >> 5;            // 0..7
  const int mBase  = (wave & 3) * 32;     // 2 M-tiles: mBase, mBase+16
  const int nBase  = (wave >> 2) * 64;    // 4 N-tiles
  const int mBlock = blockIdx.y * BM;
  const int nBlock = blockIdx.x * BN;

  // Per-thread staging footprint: one 16-half row chunk of A and of B.
  const int sRow = tid >> 1;            // 0..127
  const int sK   = (tid & 1) * 16;      // 0 or 16
  const _Float16* gA = W + (size_t)(mBlock + sRow) * K + sK;
  const _Float16* gB = X + (size_t)(nBlock + sRow) * K + sK;
  const size_t sOff = (size_t)sRow * TSTR + sK;

  v8f acc[8] = {};

  // Prologue: stage slice 0 into buffer 0.
  {
    const v8h a0 = *(const v8h*)gA, a1 = *(const v8h*)(gA + 8);
    const v8h b0 = *(const v8h*)gB, b1 = *(const v8h*)(gB + 8);
    *(v8h*)&aT[0][sOff] = a0; *(v8h*)&aT[0][sOff + 8] = a1;
    *(v8h*)&bT[0][sOff] = b0; *(v8h*)&bT[0][sOff + 8] = b1;
  }
  __syncthreads();

  const int nk = K / BK;  // 32 / 20 / 8 — always even
  auto step = [&](int buf, int s) {
    // Issue next-slice global loads; they complete under the WMMAs.
    v8h a0, a1, b0, b1;
    const bool more = (s + 1 < nk);
    if (more) {
      const _Float16* pa = gA + (size_t)(s + 1) * BK;
      const _Float16* pb = gB + (size_t)(s + 1) * BK;
      a0 = *(const v8h*)pa; a1 = *(const v8h*)(pa + 8);
      b0 = *(const v8h*)pb; b1 = *(const v8h*)(pb + 8);
    }
    if (s + 2 < nk) {  // WGP-scope prefetch of the slice after next
      __builtin_prefetch(gA + (size_t)(s + 2) * BK, 0, 3);
      __builtin_prefetch(gB + (size_t)(s + 2) * BK, 0, 3);
    }

    // Compute current slice from LDS buffer `buf`.
    const v16h fa0 = load_frag_a(&aT[buf][0], mBase, lane);
    const v16h fa1 = load_frag_a(&aT[buf][0], mBase + 16, lane);
#pragma unroll
    for (int t = 0; t < 4; ++t) {
      const v16h fb = load_frag_b(&bT[buf][0], nBase + t * 16, lane);
      acc[t] = __builtin_amdgcn_wmma_f32_16x16x32_f16(
          false, fa0, false, fb, (short)0, acc[t], false, false);
      acc[4 + t] = __builtin_amdgcn_wmma_f32_16x16x32_f16(
          false, fa1, false, fb, (short)0, acc[4 + t], false, false);
    }
    // Spill staged registers into the other buffer; the single barrier both
    // publishes them and proves this buffer's readers are done before it is
    // overwritten in the next step.
    if (more) {
      const int nb = buf ^ 1;
      *(v8h*)&aT[nb][sOff] = a0; *(v8h*)&aT[nb][sOff + 8] = a1;
      *(v8h*)&bT[nb][sOff] = b0; *(v8h*)&bT[nb][sOff + 8] = b1;
    }
    __syncthreads();
  };

  for (int s = 0; s < nk; s += 2) {
    step(0, s);
    step(1, s + 1);
  }

  // Epilogue: bias + ReLU, contiguous 16B stores into (N, M) layout.
  // C layout: VGPR r -> M = 8*(lane>>4) + r within tile; N = lane&15.
  const int mHalf = (lane >> 4) << 3;                 // 0 or 8
  const int m0 = mBlock + mBase + mHalf;              // tile 0 rows m0..m0+7
  const int col0 = nBlock + nBase + (lane & 15);
#pragma unroll
  for (int t = 0; t < 4; ++t) {
    const int c = col0 + t * 16;
#pragma unroll
    for (int half = 0; half < 2; ++half) {            // M-tile 0 / 1
      const int mrow = m0 + half * 16;
      v8h o;
#pragma unroll
      for (int r = 0; r < 8; ++r) {
        float v = acc[half * 4 + t][r] + bias[mrow + r];
        v = v > 0.0f ? v : 0.0f;
        o[r] = (_Float16)v;
      }
      *(v8h*)&Y[(size_t)c * M + mrow] = o;
    }
  }
}

// ---------------------------------------------------------------------------
// 5) Final layer: read obj, compute only the selected 4 (rot) + 3 (trans)
//    output rows from h3r/h3t (f16, N x 128). Writes d_out =
//    [ rx: N x 4 | tx: N x 3 ] fp32, channel fastest.
// ---------------------------------------------------------------------------
__global__ __launch_bounds__(256)
void final_kernel(const _Float16* __restrict__ h3r, const _Float16* __restrict__ h3t,
                  const float* __restrict__ w4r, const float* __restrict__ b4r,
                  const float* __restrict__ w4t, const float* __restrict__ b4t,
                  const int* __restrict__ obj, float* __restrict__ out) {
  __shared__ float wr[4 * L3D], wt[3 * L3D], br[4], bt[3];
  const int o = obj[0];
  for (int i = threadIdx.x; i < 4 * L3D; i += blockDim.x) wr[i] = w4r[(size_t)(o * 4) * L3D + i];
  for (int i = threadIdx.x; i < 3 * L3D; i += blockDim.x) wt[i] = w4t[(size_t)(o * 3) * L3D + i];
  if (threadIdx.x < 4) br[threadIdx.x] = b4r[o * 4 + threadIdx.x];
  if (threadIdx.x < 3) bt[threadIdx.x] = b4t[o * 3 + threadIdx.x];
  __syncthreads();

  const int n = blockIdx.x * blockDim.x + threadIdx.x;
  const _Float16* hr = h3r + (size_t)n * L3D;   // contiguous per point
  const _Float16* ht = h3t + (size_t)n * L3D;
  float ar0 = br[0], ar1 = br[1], ar2 = br[2], ar3 = br[3];
  float at0 = bt[0], at1 = bt[1], at2 = bt[2];
#pragma unroll 8
  for (int k = 0; k < L3D; ++k) {
    const float hv = (float)hr[k];
    const float tv = (float)ht[k];
    ar0 = fmaf(wr[0 * L3D + k], hv, ar0);
    ar1 = fmaf(wr[1 * L3D + k], hv, ar1);
    ar2 = fmaf(wr[2 * L3D + k], hv, ar2);
    ar3 = fmaf(wr[3 * L3D + k], hv, ar3);
    at0 = fmaf(wt[0 * L3D + k], tv, at0);
    at1 = fmaf(wt[1 * L3D + k], tv, at1);
    at2 = fmaf(wt[2 * L3D + k], tv, at2);
  }
  float* rx = out;
  float* tx = out + (size_t)4 * N_PTS;
  rx[n * 4 + 0] = ar0; rx[n * 4 + 1] = ar1; rx[n * 4 + 2] = ar2; rx[n * 4 + 3] = ar3;
  tx[n * 3 + 0] = at0; tx[n * 3 + 1] = at1; tx[n * 3 + 2] = at2;
}

// ---------------------------------------------------------------------------
// Launch
// ---------------------------------------------------------------------------
extern "C" void kernel_launch(void* const* d_in, const int* in_sizes, int n_in,
                              void* d_out, int out_size, void* d_ws, size_t ws_size,
                              hipStream_t stream) {
  (void)in_sizes; (void)n_in; (void)out_size; (void)ws_size;

  const float* emb1 = (const float*)d_in[0];
  const float* emb2 = (const float*)d_in[1];
  const float* t1   = (const float*)d_in[2];
  const float* t2   = (const float*)d_in[3];
  const int*   obj  = (const int*)d_in[4];
  const float* w1r = (const float*)d_in[5];  const float* b1r = (const float*)d_in[6];
  const float* w2r = (const float*)d_in[7];  const float* b2r = (const float*)d_in[8];
  const float* w3r = (const float*)d_in[9];  const float* b3r = (const float*)d_in[10];
  const float* w4r = (const float*)d_in[11]; const float* b4r = (const float*)d_in[12];
  const float* w1t = (const float*)d_in[13]; const float* b1t = (const float*)d_in[14];
  const float* w2t = (const float*)d_in[15]; const float* b2t = (const float*)d_in[16];
  const float* w3t = (const float*)d_in[17]; const float* b3t = (const float*)d_in[18];
  const float* w4t = (const float*)d_in[19]; const float* b4t = (const float*)d_in[20];
  float* out = (float*)d_out;

  // Bump allocator over workspace (~52 MB used).
  char* wsp = (char*)d_ws;
  auto alloc = [&](size_t bytes) -> void* {
    void* p = (void*)wsp;
    wsp += (bytes + 255) & ~(size_t)255;
    return p;
  };
  int*      idx = (int*)alloc((size_t)N_PTS * 4);
  _Float16* Xh  = (_Float16*)alloc((size_t)N_PTS * C_IN * 2);   // (N, C_IN)
  _Float16* w1rh = (_Float16*)alloc((size_t)L1D * C_IN * 2);
  _Float16* w2rh = (_Float16*)alloc((size_t)L2D * L1D * 2);
  _Float16* w3rh = (_Float16*)alloc((size_t)L3D * L2D * 2);
  _Float16* w1th = (_Float16*)alloc((size_t)L1D * C_IN * 2);
  _Float16* w2th = (_Float16*)alloc((size_t)L2D * L1D * 2);
  _Float16* w3th = (_Float16*)alloc((size_t)L3D * L2D * 2);
  _Float16* h1r = (_Float16*)alloc((size_t)N_PTS * L1D * 2);    // (N, L1)
  _Float16* h2r = (_Float16*)alloc((size_t)N_PTS * L2D * 2);    // (N, L2)
  _Float16* h3r = (_Float16*)alloc((size_t)N_PTS * L3D * 2);    // (N, L3)
  _Float16* h1t = (_Float16*)alloc((size_t)N_PTS * L1D * 2);
  _Float16* h2t = (_Float16*)alloc((size_t)N_PTS * L2D * 2);
  _Float16* h3t = (_Float16*)alloc((size_t)N_PTS * L3D * 2);

  // 1) KNN
  knn_kernel<<<N_PTS / 256, 256, 0, stream>>>(t1, t2, idx);

  // 2) weight f32 -> f16
  auto cvt = [&](const float* s, _Float16* d, int n) {
    cvt_f32_f16_kernel<<<(n + 255) / 256, 256, 0, stream>>>(s, d, n);
  };
  cvt(w1r, w1rh, L1D * C_IN);  cvt(w2r, w2rh, L2D * L1D);  cvt(w3r, w3rh, L3D * L2D);
  cvt(w1t, w1th, L1D * C_IN);  cvt(w2t, w2th, L2D * L1D);  cvt(w3t, w3th, L3D * L2D);

  // 3) gather + concat -> X (N, C_IN) f16, L2-resident
  pack_x_kernel<<<dim3(N_PTS / 64, C_IN / 64), 256, 0, stream>>>(emb1, emb2, idx, Xh);

  // 4) MLP trunks (WMMA)
  auto gemm = [&](const _Float16* W, const _Float16* Xi, const float* bias,
                  _Float16* Y, int M, int K) {
    gemm_bias_relu_wmma<<<dim3(N_PTS / BN, M / BM), 256, 0, stream>>>(W, Xi, bias, Y, M, K);
  };
  gemm(w1rh, Xh,  b1r, h1r, L1D, C_IN);
  gemm(w2rh, h1r, b2r, h2r, L2D, L1D);
  gemm(w3rh, h2r, b3r, h3r, L3D, L2D);
  gemm(w1th, Xh,  b1t, h1t, L1D, C_IN);
  gemm(w2th, h1t, b2t, h2t, L2D, L1D);
  gemm(w3th, h2t, b3t, h3t, L3D, L2D);

  // 5) per-object final rows
  final_kernel<<<N_PTS / 256, 256, 0, stream>>>(h3r, h3t, w4r, b4r, w4t, b4t, obj, out);
}
